// MultiHeadAttention_71459665871471
// MI455X (gfx1250) — matrix operand order repaired and checked by reading, not compiled, not run on hardware
//
#include <hip/hip_runtime.h>

// Problem constants (match reference)
#define Bc  4
#define Sc  2048
#define Dc  1024
#define Hc  16
#define HDc 64

typedef __attribute__((ext_vector_type(16))) __bf16        v16bf;
typedef __attribute__((ext_vector_type(8)))  float         v8f;
typedef __attribute__((ext_vector_type(8)))  unsigned short us8;
typedef __attribute__((ext_vector_type(4)))  unsigned short us4;
typedef __attribute__((ext_vector_type(4)))  float         f4;
typedef __attribute__((ext_vector_type(4)))  int           i4;

union BFrag { us8 u[2]; v16bf v; };   // 32 bytes, 16 bf16 per lane

// Native f32 -> bf16 (backend emits v_cvt_pk_bf16_f32)
__device__ __forceinline__ unsigned short f2bf(float f) {
  union { __bf16 h; unsigned short u; } cv;
  cv.h = (__bf16)f;
  return cv.u;
}

__device__ __forceinline__ v8f wmma_bf16(const BFrag& a, const BFrag& b, v8f c) {
  return __builtin_amdgcn_wmma_f32_16x16x32_bf16(false, a.v, false, b.v,
                                                 (short)0, c, false, false);
}

// generic LDS pointer: low 32 bits are the LDS byte address
__device__ __forceinline__ unsigned ldsoff(const void* p) {
  return (unsigned)(unsigned long long)p;
}

// ---- CDNA5 async global->LDS (ASYNCcnt) ----
__device__ __forceinline__ void async_copy4(void* lds, const void* g) {
  asm volatile("global_load_async_to_lds_b32 %0, %1, off"
               :: "v"(ldsoff(lds)), "v"(g) : "memory");
}
__device__ __forceinline__ void async_copy16(void* lds, const void* g) {
  asm volatile("global_load_async_to_lds_b128 %0, %1, off"
               :: "v"(ldsoff(lds)), "v"(g) : "memory");
}
__device__ __forceinline__ void wait_async0() {
  asm volatile("s_wait_asynccnt 0x0" ::: "memory");
}
__device__ __forceinline__ void wait_async1() {
  asm volatile("s_wait_asynccnt 0x1" ::: "memory");
}

// ---- CDNA5 Tensor Data Mover: descriptor-driven 2D tile load ----
#if defined(__has_builtin)
#if __has_builtin(__builtin_amdgcn_tensor_load_to_lds)
#define HAVE_TDM 1
#endif
#endif
#ifndef HAVE_TDM
#define HAVE_TDM 0
#endif

#if HAVE_TDM
typedef __attribute__((ext_vector_type(4))) unsigned int u32x4;
typedef __attribute__((ext_vector_type(8))) int          i32x8;
typedef __attribute__((ext_vector_type(4))) int          i32x4;

// 2-byte elements; tile_d0 contiguous elems per row, tile_d1 rows, row stride
// `stride_elems`; LDS rows padded: pad_interval_enc selects bytes-per-row
// (3 -> 64B rows, 4 -> 128B rows), pad_amount_enc = pad dwords - 1.
__device__ __forceinline__ void tdm_load_2d(
    unsigned lds_byte_addr, const void* gaddr,
    unsigned tile_d0, unsigned tile_d1, unsigned stride_elems,
    int pad_interval_enc, int pad_amount_enc) {
  const unsigned long long ga = (unsigned long long)gaddr;
  u32x4 g0;
  g0[0] = 1u;                                        // count=1 (valid user D#)
  g0[1] = lds_byte_addr;                             // lds_addr
  g0[2] = (unsigned)(ga & 0xFFFFFFFFu);              // global_addr[31:0]
  g0[3] = (unsigned)((ga >> 32) & 0x01FFFFFFu)       // global_addr[56:32]
        | (2u << 30);                                // type=2 ("image")
  i32x8 g1;
  unsigned w0 = (1u << 16)                           // data_size=1 -> 2 bytes
              | (1u << 20)                           // pad_enable
              | ((unsigned)pad_interval_enc << 22)
              | ((unsigned)pad_amount_enc << 25);
  g1[0] = (int)w0;
  g1[1] = (int)((tile_d0 & 0xFFFFu) << 16);          // tensor_dim0[15:0]
  g1[2] = (int)(((tile_d0 >> 16) & 0xFFFFu)          // tensor_dim0[31:16]
        | ((tile_d1 & 0xFFFFu) << 16));              // tensor_dim1[15:0]
  g1[3] = (int)(((tile_d1 >> 16) & 0xFFFFu)          // tensor_dim1[31:16]
        | ((tile_d0 & 0xFFFFu) << 16));              // tile_dim0
  g1[4] = (int)(tile_d1 & 0xFFFFu);                  // tile_dim1 (tile_dim2=0)
  g1[5] = (int)stride_elems;                         // tensor_dim0_stride[31:0]
  g1[6] = 0;
  g1[7] = 0;
  const i32x4 z4 = {0, 0, 0, 0};
  const i32x8 z8 = {0, 0, 0, 0, 0, 0, 0, 0};
  __builtin_amdgcn_tensor_load_to_lds(g0, g1, z4, z4, z8, 0);
}
#endif

// reductions across the 16-lane half-groups that hold one C-matrix row
__device__ __forceinline__ float hredmax(float x) {
#pragma unroll
  for (int m = 1; m < 16; m <<= 1) x = fmaxf(x, __shfl_xor(x, m, 32));
  return x;
}
__device__ __forceinline__ float hredsum(float x) {
#pragma unroll
  for (int m = 1; m < 16; m <<= 1) x += __shfl_xor(x, m, 32);
  return x;
}

// ---------------------------------------------------------------------------
// Kernel 1: pack int32 mask (B,S,S) -> 1 bit (mask re-read per head would be
// 1 GB of HBM traffic; packed it is 2 MB)
// ---------------------------------------------------------------------------
__global__ __launch_bounds__(256) void pack_mask_kernel(
    const int* __restrict__ mask, unsigned int* __restrict__ bits, int nwords) {
  int w = blockIdx.x * blockDim.x + threadIdx.x;
  if (w >= nwords) return;
  const int* p = mask + (size_t)w * 32;
  unsigned int out = 0u;
#pragma unroll
  for (int i = 0; i < 32; i += 4) {
    i4 q = *(const i4*)(p + i);
#pragma unroll
    for (int e = 0; e < 4; ++e)
      out |= (q[e] != 0 ? 1u : 0u) << (i + e);
  }
  bits[w] = out;
}

// ---------------------------------------------------------------------------
// Kernel 1b: bulk f32 -> bf16 (weights; done once so GEMM B-tiles are pure
// copies eligible for TDM, and W re-reads cost half the bytes)
// ---------------------------------------------------------------------------
__global__ __launch_bounds__(256) void cvt_bf16_kernel(
    const float* __restrict__ src, unsigned short* __restrict__ dst, int n8) {
  int i = blockIdx.x * blockDim.x + threadIdx.x;
  if (i >= n8) return;
  const f4* s = (const f4*)src + (size_t)i * 2;
  f4 a = s[0], b = s[1];
  us8 o;
#pragma unroll
  for (int e = 0; e < 4; ++e) { o[e] = f2bf(a[e]); o[4 + e] = f2bf(b[e]); }
  *(us8*)&dst[(size_t)i * 8] = o;
}

// ---------------------------------------------------------------------------
// Kernel 2: out = X(f32) @ Wbf^T(bf16) -> bf16, head-split layout.
//   vtrans==0 : out[b][h][s][hd]   (Q, K)
//   vtrans==1 : out[b][h][hd][s]   (V, pre-transposed for P@V B-fragments)
// 64x128 tile; B staged by TDM, A converted in-regs; 2-deep pipeline:
// tile k+1's TDM/global-loads overlap tile k's WMMAs (s_wait_tensorcnt 1).
// ---------------------------------------------------------------------------
__global__ __launch_bounds__(256) void proj_gemm_bf16(
    const float* __restrict__ X, const unsigned short* __restrict__ Wbf,
    unsigned short* __restrict__ out, int vtrans) {
  __shared__ unsigned short Ax[2][64][40];
  __shared__ unsigned short Bw[2][128][40];
  const int tid  = threadIdx.x;
  const int row0 = blockIdx.y * 64;
  const int col0 = blockIdx.x * 128;
  const int wv = tid >> 5, lane = tid & 31, lh = lane >> 4, nl = lane & 15;
  const int mbase = (wv & 1) * 32;
  const int nbase = (wv >> 1) * 32;

  auto issueB = [&](int kk, int buf) {
#if HAVE_TDM
    if (wv == 0)
      tdm_load_2d(ldsoff(&Bw[buf][0][0]), Wbf + (size_t)col0 * Dc + kk,
                  32, 128, Dc, 3, 3);
#else
#pragma unroll
    for (int i = 0; i < 2; ++i) {
      int c = tid + i * 256;
      int r = c >> 2, cc = (c & 3) * 8;
      *(us8*)&Bw[buf][r][cc] =
          *(const us8*)&Wbf[(size_t)(col0 + r) * Dc + kk + cc];
    }
#endif
  };
  auto loadA = [&](int kk, f4* xa) {
#pragma unroll
    for (int i = 0; i < 2; ++i) {
      int c = tid + i * 256;
      int r = c >> 3, cc = (c & 7) * 4;
      xa[i] = *(const f4*)&X[(size_t)(row0 + r) * Dc + kk + cc];
      __builtin_prefetch(&X[(size_t)(row0 + r) * Dc + ((kk + 32) & (Dc - 1)) + cc],
                         0, 0);
    }
  };
  auto storeA = [&](int buf, const f4* xa) {
#pragma unroll
    for (int i = 0; i < 2; ++i) {
      int c = tid + i * 256;
      int r = c >> 3, cc = (c & 7) * 4;
      us4 xb;
#pragma unroll
      for (int e = 0; e < 4; ++e) xb[e] = f2bf(xa[i][e]);
      *(us4*)&Ax[buf][r][cc] = xb;
    }
  };

  const v8f vzero = {0.f,0.f,0.f,0.f,0.f,0.f,0.f,0.f};
  v8f acc[2][2];
#pragma unroll
  for (int i = 0; i < 2; ++i)
#pragma unroll
    for (int j = 0; j < 2; ++j) acc[i][j] = vzero;

  // pipeline prologue: tile 0
  issueB(0, 0);
  {
    f4 xa0[2];
    loadA(0, xa0);
    storeA(0, xa0);
  }

  for (int k0 = 0, tix = 0; k0 < Dc; k0 += 32, ++tix) {
    const int p = tix & 1;
    const bool more = (k0 + 32) < Dc;
    f4 xa[2];
    if (more) {
      issueB(k0 + 32, p ^ 1);     // TDM for next tile runs under this compute
      loadA(k0 + 32, xa);         // global loads for next A tile in flight
    }
#if HAVE_TDM
    if (wv == 0) {
      if (more) __builtin_amdgcn_s_wait_tensorcnt(1);   // only tile k must land
      else      __builtin_amdgcn_s_wait_tensorcnt(0);
    }
#endif
    __syncthreads();
    BFrag a[2], b[2];
    const int koff = lh * 8, kb = lh * 8;
#pragma unroll
    for (int i = 0; i < 2; ++i) {
      a[i].u[0] = *(const us8*)&Ax[p][mbase + i * 16 + nl][koff];
      a[i].u[1] = *(const us8*)&Ax[p][mbase + i * 16 + nl][koff + 16];
    }
#pragma unroll
    for (int j = 0; j < 2; ++j) {
      b[j].u[0] = *(const us8*)&Bw[p][nbase + j * 16 + nl][kb];
      b[j].u[1] = *(const us8*)&Bw[p][nbase + j * 16 + nl][kb + 16];
    }
#pragma unroll
    for (int i = 0; i < 2; ++i)
#pragma unroll
      for (int j = 0; j < 2; ++j) acc[i][j] = wmma_bf16(a[i], b[j], acc[i][j]);
    if (more) storeA(p ^ 1, xa);  // convert+store after compute issued
    __syncthreads();
  }
#pragma unroll
  for (int i = 0; i < 2; ++i)
#pragma unroll
    for (int j = 0; j < 2; ++j) {
      const int colg = col0 + nbase + j * 16 + nl;
      const int hh = colg >> 6, hd = colg & 63;
#pragma unroll
      for (int v = 0; v < 8; ++v) {
        const int rowg = row0 + mbase + i * 16 + v + lh * 8;
        const int bb = rowg >> 11, ss = rowg & (Sc - 1);
        const size_t idx = vtrans
            ? (((size_t)bb * Hc + hh) * HDc + hd) * Sc + ss
            : (((size_t)bb * Hc + hh) * Sc + ss) * HDc + hd;
        out[idx] = f2bf(acc[i][j][v]);
      }
    }
}

// ---------------------------------------------------------------------------
// Kernel 3: flash attention.  Block = 4 waves = 64 query rows of one (b,h).
// K/Vt tiles via TDM, mask via async-LDS, 2-deep pipeline: tile t+1's DMA
// overlaps tile t's WMMAs (s_wait_tensorcnt 2 / s_wait_asynccnt 1).
// ---------------------------------------------------------------------------
__global__ __launch_bounds__(128) void flash_attn_kernel(
    const unsigned short* __restrict__ Qb, const unsigned short* __restrict__ Kb,
    const unsigned short* __restrict__ Vtg, const unsigned int* __restrict__ Mbits,
    unsigned short* __restrict__ Ctx) {
  __shared__ unsigned short Kt[2][64][72]; // [key][hd]  -> B-frags for Q@K^T
  __shared__ unsigned short Vt[2][64][72]; // [hd][key]  -> B-frags for P@V
  __shared__ unsigned short Pl[4][16][72]; // per-wave P tile (C->A relayout)
  __shared__ unsigned int   Mw[2][64][2];  // mask bits per key tile

  const int tid = threadIdx.x;
  const int bz = blockIdx.z, h = blockIdx.y, q0 = blockIdx.x * 64;
  const int wv = tid >> 5, lane = tid & 31, lh = lane >> 4, nl = lane & 15;
  const size_t hb  = ((size_t)bz * Hc + h) * Sc * HDc;   // K base  [key][hd]
  const size_t vtb = ((size_t)bz * Hc + h) * HDc * Sc;   // Vt base [hd][key]

  auto issue_tile = [&](int tt, int buf) {
    const int kt0 = tt * 64;
#if HAVE_TDM
    if (wv == 0) {   // two descriptors move both 64x64 bf16 tiles
      tdm_load_2d(ldsoff(&Kt[buf][0][0]), Kb + hb + (size_t)kt0 * HDc,
                  64, 64, HDc, 4, 3);
      tdm_load_2d(ldsoff(&Vt[buf][0][0]), Vtg + vtb + kt0,
                  64, 64, Sc, 4, 3);
    }
#else
#pragma unroll
    for (int i = 0; i < 4; ++i) {
      int c = tid + i * 128;
      int r = c >> 3, cc = (c & 7) * 8;
      async_copy16(&Kt[buf][r][cc], Kb + hb + (size_t)(kt0 + r) * HDc + cc);
      async_copy16(&Vt[buf][r][cc], Vtg + vtb + (size_t)r * Sc + kt0 + cc);
    }
#endif
    int r = tid >> 1, wi = tid & 1;
    async_copy4(&Mw[buf][r][wi],
                &Mbits[((size_t)bz * Sc + q0 + r) * (Sc / 32) + tt * 2 + wi]);
  };

  BFrag qa0, qa1;                           // Q rows held in VGPRs throughout
  {
    const int koff = lh * 8;
    const unsigned short* qp = Qb + hb + (size_t)(q0 + wv * 16 + nl) * HDc;
    qa0.u[0] = *(const us8*)(qp + koff);
    qa0.u[1] = *(const us8*)(qp + koff + 16);
    qa1.u[0] = *(const us8*)(qp + 32 + koff);
    qa1.u[1] = *(const us8*)(qp + 32 + koff + 16);
  }

  float mrun[8], lrun[8];
#pragma unroll
  for (int v = 0; v < 8; ++v) { mrun[v] = -1e30f; lrun[v] = 0.f; }
  const v8f vzero = {0.f,0.f,0.f,0.f,0.f,0.f,0.f,0.f};
  v8f o[4];
#pragma unroll
  for (int n = 0; n < 4; ++n) o[n] = vzero;

  issue_tile(0, 0);                         // pipeline prologue

  for (int t = 0; t < Sc / 64; ++t) {
    const int p = t & 1;
    const bool more = (t + 1) < (Sc / 64);
    if (more) issue_tile(t + 1, p ^ 1);     // DMA for t+1 under compute of t
#if HAVE_TDM
    if (wv == 0) {
      if (more) __builtin_amdgcn_s_wait_tensorcnt(2);  // t's pair has landed
      else      __builtin_amdgcn_s_wait_tensorcnt(0);
    }
    if (more) wait_async1(); else wait_async0();       // mask word for t
#else
    wait_async0();
#endif
    __syncthreads();

    // ---- scores: S = Q(16x64) @ K^T -> 4 accsets of 16x16, K-dim=64 ----
    v8f sc[4];
#pragma unroll
    for (int j = 0; j < 4; ++j) {
      const int key = j * 16 + nl;
      BFrag b0, b1;
      b0.u[0] = *(const us8*)&Kt[p][key][lh * 8];
      b0.u[1] = *(const us8*)&Kt[p][key][lh * 8 + 16];
      b1.u[0] = *(const us8*)&Kt[p][key][32 + lh * 8];
      b1.u[1] = *(const us8*)&Kt[p][key][32 + lh * 8 + 16];
      v8f s = vzero;
      s = wmma_bf16(qa0, b0, s);
      s = wmma_bf16(qa1, b1, s);
      sc[j] = s;
    }

    // ---- online softmax (fp32), write P (bf16) to per-wave LDS ----
    float corr_row[8];
#pragma unroll
    for (int v = 0; v < 8; ++v) {
      const int qrl = wv * 16 + v + lh * 8;
      const unsigned int w0 = Mw[p][qrl][0], w1 = Mw[p][qrl][1];
      unsigned int ok[4];
      ok[0] = (w0 >> nl) & 1u;  ok[1] = (w0 >> (16 + nl)) & 1u;
      ok[2] = (w1 >> nl) & 1u;  ok[3] = (w1 >> (16 + nl)) & 1u;
      float s[4], rmax = -1e30f;
#pragma unroll
      for (int j = 0; j < 4; ++j) {
        float x = sc[j][v] * 0.125f;        // 1/sqrt(HD)
        s[j] = ok[j] ? x : -1e30f;
        rmax = fmaxf(rmax, s[j]);
      }
      rmax = hredmax(rmax);
      const float mnew = fmaxf(mrun[v], rmax);
      const float corr = __expf(mrun[v] - mnew);
      float psum = 0.f;
#pragma unroll
      for (int j = 0; j < 4; ++j) {
        float pj = ok[j] ? __expf(s[j] - mnew) : 0.f;
        psum += pj;
        Pl[wv][v + lh * 8][j * 16 + nl] = f2bf(pj);
      }
      lrun[v] = lrun[v] * corr + hredsum(psum);
      mrun[v] = mnew;
      corr_row[v] = corr;
    }
#pragma unroll
    for (int n = 0; n < 4; ++n)
#pragma unroll
      for (int v = 0; v < 8; ++v) o[n][v] *= corr_row[v];

    // ---- O += P(16x64) @ V(64x64) ----
    BFrag pa0, pa1;
    {
      const int koff = lh * 8;
      pa0.u[0] = *(const us8*)&Pl[wv][nl][koff];
      pa0.u[1] = *(const us8*)&Pl[wv][nl][koff + 16];
      pa1.u[0] = *(const us8*)&Pl[wv][nl][32 + koff];
      pa1.u[1] = *(const us8*)&Pl[wv][nl][32 + koff + 16];
    }
#pragma unroll
    for (int n = 0; n < 4; ++n) {
      const int hd = n * 16 + nl;
      BFrag b0, b1;
      b0.u[0] = *(const us8*)&Vt[p][hd][lh * 8];
      b0.u[1] = *(const us8*)&Vt[p][hd][lh * 8 + 16];
      b1.u[0] = *(const us8*)&Vt[p][hd][32 + lh * 8];
      b1.u[1] = *(const us8*)&Vt[p][hd][32 + lh * 8 + 16];
      o[n] = wmma_bf16(pa0, b0, o[n]);
      o[n] = wmma_bf16(pa1, b1, o[n]);
    }
    __syncthreads();
  }

  // ---- normalize and store ctx (bf16, [B][S][D] with d = h*HD+hd) ----
#pragma unroll
  for (int n = 0; n < 4; ++n)
#pragma unroll
    for (int v = 0; v < 8; ++v) {
      const int srow = q0 + wv * 16 + v + lh * 8;
      const float val = o[n][v] / fmaxf(lrun[v], 1e-30f);
      Ctx[((size_t)bz * Sc + srow) * Dc + h * HDc + n * 16 + nl] = f2bf(val);
    }
}

// ---------------------------------------------------------------------------
// Kernel 4: out(f32) = ctx(bf16) @ Wo^T(bf16).  Both tiles by TDM, 2-deep
// pipeline (s_wait_tensorcnt 2).
// ---------------------------------------------------------------------------
__global__ __launch_bounds__(256) void out_gemm_f32(
    const unsigned short* __restrict__ Xb, const unsigned short* __restrict__ Wbf,
    float* __restrict__ out) {
  __shared__ unsigned short Ax[2][64][40];
  __shared__ unsigned short Bw[2][128][40];
  const int tid  = threadIdx.x;
  const int row0 = blockIdx.y * 64;
  const int col0 = blockIdx.x * 128;
  const int wv = tid >> 5, lane = tid & 31, lh = lane >> 4, nl = lane & 15;
  const int mbase = (wv & 1) * 32;
  const int nbase = (wv >> 1) * 32;

  auto issue_tiles = [&](int kk, int buf) {
#if HAVE_TDM
    if (wv == 0) {
      tdm_load_2d(ldsoff(&Ax[buf][0][0]), Xb  + (size_t)row0 * Dc + kk,
                  32, 64, Dc, 3, 3);
      tdm_load_2d(ldsoff(&Bw[buf][0][0]), Wbf + (size_t)col0 * Dc + kk,
                  32, 128, Dc, 3, 3);
    }
#else
    {
      int r = tid >> 2, cc = (tid & 3) * 8;
      async_copy16(&Ax[buf][r][cc], &Xb[(size_t)(row0 + r) * Dc + kk + cc]);
    }
#pragma unroll
    for (int i = 0; i < 2; ++i) {
      int c = tid + i * 256;
      int r = c >> 2, cc = (c & 3) * 8;
      *(us8*)&Bw[buf][r][cc] =
          *(const us8*)&Wbf[(size_t)(col0 + r) * Dc + kk + cc];
    }
#endif
  };

  const v8f vzero = {0.f,0.f,0.f,0.f,0.f,0.f,0.f,0.f};
  v8f acc[2][2];
#pragma unroll
  for (int i = 0; i < 2; ++i)
#pragma unroll
    for (int j = 0; j < 2; ++j) acc[i][j] = vzero;

  issue_tiles(0, 0);                        // pipeline prologue

  for (int k0 = 0, tix = 0; k0 < Dc; k0 += 32, ++tix) {
    const int p = tix & 1;
    const bool more = (k0 + 32) < Dc;
    if (more) issue_tiles(k0 + 32, p ^ 1);
#if HAVE_TDM
    if (wv == 0) {
      if (more) __builtin_amdgcn_s_wait_tensorcnt(2);
      else      __builtin_amdgcn_s_wait_tensorcnt(0);
    }
#else
    wait_async0();
#endif
    __syncthreads();
    BFrag a[2], b[2];
    const int koff = lh * 8, kb = lh * 8;
#pragma unroll
    for (int i = 0; i < 2; ++i) {
      a[i].u[0] = *(const us8*)&Ax[p][mbase + i * 16 + nl][koff];
      a[i].u[1] = *(const us8*)&Ax[p][mbase + i * 16 + nl][koff + 16];
    }
#pragma unroll
    for (int j = 0; j < 2; ++j) {
      b[j].u[0] = *(const us8*)&Bw[p][nbase + j * 16 + nl][kb];
      b[j].u[1] = *(const us8*)&Bw[p][nbase + j * 16 + nl][kb + 16];
    }
#pragma unroll
    for (int i = 0; i < 2; ++i)
#pragma unroll
      for (int j = 0; j < 2; ++j) acc[i][j] = wmma_bf16(a[i], b[j], acc[i][j]);
    __syncthreads();
  }
#pragma unroll
  for (int i = 0; i < 2; ++i)
#pragma unroll
    for (int j = 0; j < 2; ++j) {
      const int colg = col0 + nbase + j * 16 + nl;
#pragma unroll
      for (int v = 0; v < 8; ++v) {
        const int rowg = row0 + mbase + i * 16 + v + lh * 8;
        out[(size_t)rowg * Dc + colg] = acc[i][j][v];
      }
    }
}

// ---------------------------------------------------------------------------
extern "C" void kernel_launch(void* const* d_in, const int* in_sizes, int n_in,
                              void* d_out, int out_size, void* d_ws, size_t ws_size,
                              hipStream_t stream) {
  (void)in_sizes; (void)n_in; (void)out_size; (void)ws_size;
  const float* q    = (const float*)d_in[0];
  const float* k    = (const float*)d_in[1];
  const float* v    = (const float*)d_in[2];
  const int*   mask = (const int*)  d_in[3];
  const float* Wq   = (const float*)d_in[4];
  const float* Wk   = (const float*)d_in[5];
  const float* Wv   = (const float*)d_in[6];
  const float* Wo   = (const float*)d_in[7];
  float* out = (float*)d_out;

  const size_t qkvElems = (size_t)Bc * Hc * Sc * HDc;   // 8.4M bf16
  const size_t wElems   = (size_t)Dc * Dc;              // 1M bf16 per W
  unsigned short* Qw  = (unsigned short*)d_ws;
  unsigned short* Kw  = Qw + qkvElems;
  unsigned short* Vtw = Kw + qkvElems;                  // [B][H][HD][S]
  unsigned short* Cw  = Vtw + qkvElems;                 // ctx, B*S*D bf16
  unsigned short* Wqb = Cw + (size_t)Bc * Sc * Dc;
  unsigned short* Wkb = Wqb + wElems;
  unsigned short* Wvb = Wkb + wElems;
  unsigned short* Wob = Wvb + wElems;
  unsigned int*   Mb  = (unsigned int*)(Wob + wElems);

  const int nwords = Bc * Sc * (Sc / 32);
  pack_mask_kernel<<<nwords / 256, 256, 0, stream>>>(mask, Mb, nwords);

  const int n8 = (int)(wElems / 8);
  cvt_bf16_kernel<<<n8 / 256, 256, 0, stream>>>(Wq, Wqb, n8);
  cvt_bf16_kernel<<<n8 / 256, 256, 0, stream>>>(Wk, Wkb, n8);
  cvt_bf16_kernel<<<n8 / 256, 256, 0, stream>>>(Wv, Wvb, n8);
  cvt_bf16_kernel<<<n8 / 256, 256, 0, stream>>>(Wo, Wob, n8);

  dim3 ggrid(Dc / 128, (Bc * Sc) / 64);
  proj_gemm_bf16<<<ggrid, 256, 0, stream>>>(q, Wqb, Qw, 0);
  proj_gemm_bf16<<<ggrid, 256, 0, stream>>>(k, Wkb, Kw, 0);
  proj_gemm_bf16<<<ggrid, 256, 0, stream>>>(v, Wvb, Vtw, 1);

  flash_attn_kernel<<<dim3(Sc / 64, Hc, Bc), 128, 0, stream>>>(Qw, Kw, Vtw, Mb, Cw);

  out_gemm_f32<<<ggrid, 256, 0, stream>>>(Cw, Wob, out);
}
